// MoELayer_68049461838426
// MI455X (gfx1250) — matrix operand order repaired and checked by reading, not compiled, not run on hardware
//
#include <hip/hip_runtime.h>
#include <hip/hip_bf16.h>
#include <cstdint>

// ---------------- problem constants ----------------
#define E_      8
#define KTOP_   2
#define D_      768
#define H_      3072
#define T_      2048        // B*S
#define TILE_M  32
#define LB_SCALE 0.01f

typedef __attribute__((ext_vector_type(16))) __bf16 v16bf;
typedef __attribute__((ext_vector_type(8)))  __bf16 v8bf;
typedef __attribute__((ext_vector_type(8)))  float  v8f;

// ---------------- CDNA5 async global->LDS copy (guarded) ----------------
#if defined(__gfx1250__) && __has_builtin(__builtin_amdgcn_global_load_async_to_lds_b128)
  #define ASYNC_CP 1
#else
  #define ASYNC_CP 0
#endif

// builtin prototype (from hipcc diagnostic): param0 = <4 x i32> addrspace(1)*,
// param1 = <4 x i32> addrspace(3)*, then imm offset + imm cpol.
typedef int v4i __attribute__((vector_size(16)));
typedef __attribute__((address_space(1))) v4i gv4i;
typedef __attribute__((address_space(3))) v4i lv4i;

__device__ __forceinline__ void cp16_async(const __bf16* g, __bf16* l) {
#if ASYNC_CP
  __builtin_amdgcn_global_load_async_to_lds_b128((gv4i*)g, (lv4i*)l, 0, 0);
#else
  *(v8bf*)l = *(const v8bf*)g;      // fallback: plain 16B copy
#endif
}

#if ASYNC_CP
  #if __has_builtin(__builtin_amdgcn_s_wait_asynccnt)
    #define ASYNC_WAIT(n) __builtin_amdgcn_s_wait_asynccnt(n)
  #else
    #define ASYNC_WAIT(n) asm volatile("s_wait_asynccnt %0" :: "i"(n) : "memory")
  #endif
#else
  #define ASYNC_WAIT(n) ((void)0)
#endif

// ---------------- WMMA fragment helpers (wave32, ISA 7.12.2) ----------------
__device__ __forceinline__ v16bf combine16(v8bf lo, v8bf hi) {
  return __builtin_shufflevector(lo, hi, 0,1,2,3,4,5,6,7,8,9,10,11,12,13,14,15);
}

// A 16x32 bf16 frag from row-major LDS panel [M][K] (stride in halves).
__device__ __forceinline__ v16bf load_a_frag(const __bf16* p, int stride,
                                             int m0, int k0, int lane) {
  const __bf16* row = p + (size_t)(m0 + (lane & 15)) * stride + k0 + ((lane >> 4) << 3);
  v8bf lo = *(const v8bf*)row;
  v8bf hi = *(const v8bf*)(row + 16);
  return combine16(lo, hi);
}

// B 32x16 bf16 frag from K-transposed LDS staging [n][k], k-stride 32.
__device__ __forceinline__ v16bf load_b_frag(const __bf16* p, int n0, int lane) {
  const __bf16* row = p + (size_t)(n0 + (lane & 15)) * 32 + ((lane >> 4) << 3);
  v8bf lo = *(const v8bf*)row;
  v8bf hi = *(const v8bf*)(row + 16);
  return combine16(lo, hi);
}

#define WMMA_BF16(A, B, C) \
  __builtin_amdgcn_wmma_f32_16x16x32_bf16(false, (A), false, (B), (short)0, (C), false, false)

// ---- staging: issue async copies for a 128(n) x 32(k) bf16 tile, [n][k] LDS ----
__device__ __forceinline__ void stage_one(const __bf16* g, __bf16* l,
                                          int n0, int k0, int tid, int strideK) {
  #pragma unroll
  for (int j = 0; j < 2; ++j) {                      // 512 chunks / 256 thr
    int idx = tid + (j << 8);
    int nn = idx >> 2, c = (idx & 3) << 3;
    cp16_async(g + (size_t)(n0 + nn) * strideK + k0 + c, l + nn * 32 + c);
  }
}
__device__ __forceinline__ void stage_two(const __bf16* g1, const __bf16* g2,
                                          __bf16* l1, __bf16* l2,
                                          int n0, int k0, int tid, int strideK) {
  #pragma unroll
  for (int j = 0; j < 2; ++j) {
    int idx = tid + (j << 8);
    int nn = idx >> 2, c = (idx & 3) << 3;
    size_t go = (size_t)(n0 + nn) * strideK + k0 + c;
    cp16_async(g1 + go, l1 + nn * 32 + c);
    cp16_async(g2 + go, l2 + nn * 32 + c);
  }
}

// ---------------- kernel 0: init (zero out + counters, optional x->bf16) -----
__global__ void moe_init(float* __restrict__ out, int* __restrict__ counts,
                         float* __restrict__ gmean, const float* __restrict__ x,
                         __bf16* __restrict__ xb) {
  unsigned i = blockIdx.x * blockDim.x + threadIdx.x;
  if (i < (unsigned)(T_ * D_)) {
    out[i] = 0.0f;
    if (xb) xb[i] = (__bf16)x[i];
  }
  if (i < E_) { counts[i] = 0; gmean[i] = 0.0f; }
}

// ---------------- kernel 0b: fp32 -> bf16 transposed weight repack ----------
// src: [z][R][C] fp32 row-major ; dst: [z][C][R] bf16 row-major
__global__ void moe_repack(const float* __restrict__ src_base,
                           __bf16* __restrict__ dst_base, int R, int C) {
  __shared__ float tile[32][33];
  const float* src = src_base + (size_t)blockIdx.z * R * C;
  __bf16*      dst = dst_base + (size_t)blockIdx.z * R * C;
  int c0 = blockIdx.x * 32, r0 = blockIdx.y * 32;
  int lx = threadIdx.x & 31, ly = threadIdx.x >> 5;
  #pragma unroll
  for (int rr = ly; rr < 32; rr += 8)
    tile[rr][lx] = src[(size_t)(r0 + rr) * C + c0 + lx];
  __syncthreads();
  #pragma unroll
  for (int cc = ly; cc < 32; cc += 8)
    dst[(size_t)(c0 + cc) * R + r0 + lx] = (__bf16)tile[lx][cc];
}

// ---------------- kernel 1: gate (one wave32 per token) ----------------
__global__ void moe_gate(const float* __restrict__ x, const float* __restrict__ noise,
                         const float* __restrict__ gate_w, const float* __restrict__ nw,
                         int* __restrict__ counts, float* __restrict__ gmean,
                         int* __restrict__ toklist, float* __restrict__ gatelist,
                         int* __restrict__ ids_out) {
  const int wave = threadIdx.x >> 5;
  const int lane = threadIdx.x & 31;
  const int t = blockIdx.x * 8 + wave;
  if (t >= T_) return;

  float acc[E_];
  #pragma unroll
  for (int e = 0; e < E_; ++e) acc[e] = 0.0f;

  const float* xr = x + (size_t)t * D_;
  for (int d = lane; d < D_; d += 32) {
    float xv = xr[d];
    #pragma unroll
    for (int e = 0; e < E_; ++e) acc[e] += xv * gate_w[d * E_ + e];
  }
  #pragma unroll
  for (int e = 0; e < E_; ++e) {
    #pragma unroll
    for (int off = 16; off > 0; off >>= 1)
      acc[e] += __shfl_xor(acc[e], off, 32);
  }

  if (lane == 0) {
    float mx = acc[0];
    #pragma unroll
    for (int e = 1; e < E_; ++e) mx = fmaxf(mx, acc[e]);
    float pe[E_], s = 0.0f;
    #pragma unroll
    for (int e = 0; e < E_; ++e) { pe[e] = __expf(acc[e] - mx); s += pe[e]; }
    float inv = 1.0f / s;
    #pragma unroll
    for (int e = 0; e < E_; ++e) atomicAdd(&gmean[e], pe[e] * inv);

    float ln[E_];
    #pragma unroll
    for (int e = 0; e < E_; ++e) ln[e] = acc[e] + noise[t * E_ + e] * nw[e];
    int i0 = 0; float v0 = ln[0];
    #pragma unroll
    for (int e = 1; e < E_; ++e) if (ln[e] > v0) { v0 = ln[e]; i0 = e; }
    int i1 = (i0 == 0) ? 1 : 0; float v1 = ln[i1];
    #pragma unroll
    for (int e = 0; e < E_; ++e)
      if (e != i0 && ln[e] > v1) { v1 = ln[e]; i1 = e; }

    float g0 = 1.0f / (1.0f + __expf(v1 - v0));
    float g1 = 1.0f - g0;

    ids_out[t * KTOP_ + 0] = i0;
    ids_out[t * KTOP_ + 1] = i1;

    int p0 = atomicAdd(&counts[i0], 1);
    toklist[i0 * T_ + p0] = t; gatelist[i0 * T_ + p0] = g0;
    int p1 = atomicAdd(&counts[i1], 1);
    toklist[i1 * T_ + p1] = t; gatelist[i1 * T_ + p1] = g1;
  }
}

// ---------------- kernel 2: LB loss finalize ----------------
__global__ void moe_lb(const float* __restrict__ gmean, float* __restrict__ loss) {
  if (threadIdx.x == 0 && blockIdx.x == 0) {
    float s = 0.0f;
    #pragma unroll
    for (int e = 0; e < E_; ++e) {
      float p = gmean[e] * (1.0f / (float)T_);
      float d = p - (1.0f / (float)E_);
      s += d * d;
    }
    loss[0] = (s / (float)E_) * LB_SCALE;
  }
}

// ---------------- kernel 3: fused expert SwiGLU (bf16 WMMA) ----------------
// PRE=true : weights pre-converted to bf16 and K-transposed in ws; staging is
//            double-buffered GLOBAL_LOAD_ASYNC_TO_LDS_B128 overlapped with WMMA.
// PRE=false: in-kernel fp32->bf16 conversion staging (ws too small for repack).
// LDS: sA 48KB | sAct 192KB | 4x 8KB staging = 272KB (<=320KB per WGP).
template <bool PRE>
__global__ void __launch_bounds__(256, 1)
moe_expert(const float* __restrict__ x, const __bf16* __restrict__ xb,
           const float* __restrict__ w1f, const float* __restrict__ w2f,
           const float* __restrict__ wpf,
           const __bf16* __restrict__ w1t, const __bf16* __restrict__ w2t,
           const __bf16* __restrict__ wpt,
           const float* __restrict__ b1, const float* __restrict__ b2,
           const float* __restrict__ bp,
           const int* __restrict__ counts, const int* __restrict__ toklist,
           const float* __restrict__ gatelist, float* __restrict__ out) {
  extern __shared__ __bf16 smem[];
  __bf16* sA   = smem;                        // [32][768]
  __bf16* sAct = sA + TILE_M * D_;            // [32][3072]
  __bf16* sBh0 = sAct + TILE_M * H_;          // [128][32]
  __bf16* sBh1 = sBh0 + 4096;
  __bf16* sBg0 = sBh1 + 4096;
  __bf16* sBg1 = sBg0 + 4096;

  const int e    = blockIdx.y;
  const int row0 = blockIdx.x * TILE_M;
  const int cnt  = counts[e];
  if (row0 >= cnt) return;                    // uniform early exit
  const int rows = min(TILE_M, cnt - row0);

  const int tid  = threadIdx.x;
  const int wave = tid >> 5;
  const int lane = tid & 31;

  const int*   tl = toklist  + e * T_ + row0;
  const float* gl = gatelist + e * T_ + row0;

  // ---- gathered A panel (bf16) ----
  if constexpr (PRE) {
    for (int i = tid; i < TILE_M * (D_ / 8); i += 256) {     // 16B chunks
      int m = i / (D_ / 8), c = (i - m * (D_ / 8)) << 3;
      int t = tl[(m < rows) ? m : 0];
      cp16_async(xb + (size_t)t * D_ + c, sA + (size_t)m * D_ + c);
    }
    // completion folded into first staged-tile wait below
  } else {
    for (int i = tid; i < TILE_M * D_; i += 256) {
      int m = i / D_, d = i - m * D_;
      int t = tl[(m < rows) ? m : 0];
      sA[(size_t)m * D_ + d] = (__bf16)x[(size_t)t * D_ + d];
    }
    __syncthreads();
  }

  const __bf16* w1te = PRE ? w1t + (size_t)e * D_ * H_ : nullptr;
  const __bf16* w2te = PRE ? w2t + (size_t)e * D_ * H_ : nullptr;
  const __bf16* wpte = PRE ? wpt + (size_t)e * H_ * D_ : nullptr;
  const float*  w1fe = w1f + (size_t)e * D_ * H_;
  const float*  w2fe = w2f + (size_t)e * D_ * H_;
  const float*  wpfe = wpf + (size_t)e * H_ * D_;

  // ========== phase 1: h = xW1+b1, g = xW2+b2, act = h*silu(g) ==========
  for (int nc = 0; nc < H_ / 128; ++nc) {
    const int n0 = nc * 128;
    v8f accH0 = {}, accH1 = {}, accG0 = {}, accG1 = {};
    if constexpr (PRE)                       // prologue: tile 0 -> buffer 0
      stage_two(w1te, w2te, sBh0, sBg0, n0, 0, tid, D_);
    for (int kc = 0; kc < D_ / 32; ++kc) {
      const int k0 = kc * 32;
      if constexpr (PRE) {
        if (kc + 1 < D_ / 32) {              // issue next tile into other buffer
          stage_two(w1te, w2te, (kc & 1) ? sBh0 : sBh1, (kc & 1) ? sBg0 : sBg1,
                    n0, k0 + 32, tid, D_);
          ASYNC_WAIT(4);                     // tile kc (and A panel) complete
        } else {
          ASYNC_WAIT(0);
        }
        __syncthreads();
      } else {
        for (int i = tid; i < 128 * 32; i += 256) {
          int nn = i & 127, kk = i >> 7;     // coalesced over n
          size_t g = (size_t)(k0 + kk) * H_ + n0 + nn;
          sBh0[nn * 32 + kk] = (__bf16)w1fe[g];
          sBg0[nn * 32 + kk] = (__bf16)w2fe[g];
        }
        __syncthreads();
      }
      const __bf16* bhp = (PRE && (kc & 1)) ? sBh1 : sBh0;
      const __bf16* bgp = (PRE && (kc & 1)) ? sBg1 : sBg0;
      v16bf bh = load_b_frag(bhp, wave * 16, lane);
      v16bf bg = load_b_frag(bgp, wave * 16, lane);
      v16bf a0 = load_a_frag(sA, D_,  0, k0, lane);
      v16bf a1 = load_a_frag(sA, D_, 16, k0, lane);
      accH0 = WMMA_BF16(a0, bh, accH0);
      accH1 = WMMA_BF16(a1, bh, accH1);
      accG0 = WMMA_BF16(a0, bg, accG0);
      accG1 = WMMA_BF16(a1, bg, accG1);
      __syncthreads();
    }
    const int col = n0 + wave * 16 + (lane & 15);
    const float b1v = b1[e * H_ + col];
    const float b2v = b2[e * H_ + col];
    const int rbase = (lane >> 4) << 3;
    #pragma unroll
    for (int r = 0; r < 8; ++r) {
      {
        float h = accH0[r] + b1v, g = accG0[r] + b2v;
        float a = h * g / (1.0f + __expf(-g));
        sAct[(size_t)(r + rbase) * H_ + col] = (__bf16)a;
      }
      {
        float h = accH1[r] + b1v, g = accG1[r] + b2v;
        float a = h * g / (1.0f + __expf(-g));
        sAct[(size_t)(16 + r + rbase) * H_ + col] = (__bf16)a;
      }
    }
  }
  __syncthreads();

  // ========== phase 2: y = act @ Wp + bp, gated scatter-add ==========
  for (int ng = 0; ng < D_ / 128; ++ng) {
    const int n0 = ng * 128;
    v8f accO0 = {}, accO1 = {};
    if constexpr (PRE)
      stage_one(wpte, sBh0, n0, 0, tid, H_);
    for (int kc = 0; kc < H_ / 32; ++kc) {
      const int k0 = kc * 32;
      if constexpr (PRE) {
        if (kc + 1 < H_ / 32) {
          stage_one(wpte, (kc & 1) ? sBh0 : sBh1, n0, k0 + 32, tid, H_);
          ASYNC_WAIT(2);
        } else {
          ASYNC_WAIT(0);
        }
        __syncthreads();
      } else {
        for (int i = tid; i < 128 * 32; i += 256) {
          int nn = i & 127, kk = i >> 7;
          sBh0[nn * 32 + kk] = (__bf16)wpfe[(size_t)(k0 + kk) * D_ + n0 + nn];
        }
        __syncthreads();
      }
      const __bf16* bpnt = (PRE && (kc & 1)) ? sBh1 : sBh0;
      v16bf b  = load_b_frag(bpnt, wave * 16, lane);
      v16bf a0 = load_a_frag(sAct, H_,  0, k0, lane);
      v16bf a1 = load_a_frag(sAct, H_, 16, k0, lane);
      accO0 = WMMA_BF16(a0, b, accO0);
      accO1 = WMMA_BF16(a1, b, accO1);
      __syncthreads();
    }
    const int col = n0 + wave * 16 + (lane & 15);
    const float bpv = bp[e * D_ + col];
    const int rbase = (lane >> 4) << 3;
    #pragma unroll
    for (int r = 0; r < 8; ++r) {
      int row = r + rbase;
      if (row < rows)
        atomicAdd(&out[(size_t)tl[row] * D_ + col], gl[row] * (accO0[r] + bpv));
      int row1 = row + 16;
      if (row1 < rows)
        atomicAdd(&out[(size_t)tl[row1] * D_ + col], gl[row1] * (accO1[r] + bpv));
    }
  }
}

// ---------------- launch ----------------
extern "C" void kernel_launch(void* const* d_in, const int* in_sizes, int n_in,
                              void* d_out, int out_size, void* d_ws, size_t ws_size,
                              hipStream_t stream) {
  (void)in_sizes; (void)n_in; (void)out_size;
  const float* x   = (const float*)d_in[0];
  const float* nz  = (const float*)d_in[1];
  const float* gw  = (const float*)d_in[2];
  const float* nw  = (const float*)d_in[3];
  const float* w1  = (const float*)d_in[4];
  const float* b1  = (const float*)d_in[5];
  const float* w2  = (const float*)d_in[6];
  const float* b2  = (const float*)d_in[7];
  const float* wp  = (const float*)d_in[8];
  const float* bp  = (const float*)d_in[9];

  float* out  = (float*)d_out;                           // [T, D]
  int*   ids  = (int*)((float*)d_out + (size_t)T_ * D_); // [T, K] int32 bits
  float* loss = (float*)d_out + (size_t)T_ * D_ + (size_t)T_ * KTOP_;

  // workspace layout
  const size_t WMAT     = (size_t)E_ * D_ * H_;          // elements per weight set
  const size_t OFF_TOK  = 128;
  const size_t OFF_GATE = OFF_TOK + (size_t)E_ * T_ * sizeof(int);
  const size_t OFF_XB   = 135168;                        // 128B aligned
  const size_t OFF_W1T  = 4u * 1024u * 1024u;
  const size_t OFF_W2T  = OFF_W1T + WMAT * sizeof(__bf16);
  const size_t OFF_WPT  = OFF_W2T + WMAT * sizeof(__bf16);
  const size_t NEED     = OFF_WPT + WMAT * sizeof(__bf16);  // ~112 MB

  int*    counts   = (int*)d_ws;
  float*  gmean    = (float*)((char*)d_ws + 64);
  int*    toklist  = (int*)((char*)d_ws + OFF_TOK);
  float*  gatelist = (float*)((char*)d_ws + OFF_GATE);
  __bf16* xb       = (__bf16*)((char*)d_ws + OFF_XB);
  __bf16* w1t      = (__bf16*)((char*)d_ws + OFF_W1T);
  __bf16* w2t      = (__bf16*)((char*)d_ws + OFF_W2T);
  __bf16* wpt      = (__bf16*)((char*)d_ws + OFF_WPT);

  const bool pre = (ws_size >= NEED);

  // 0) init out + counters (+ x -> bf16 when pre-pack path active)
  moe_init<<<(T_ * D_ + 255) / 256, 256, 0, stream>>>(out, counts, gmean, x,
                                                      pre ? xb : nullptr);
  // 1) gate
  moe_gate<<<T_ / 8, 256, 0, stream>>>(x, nz, gw, nw, counts, gmean,
                                       toklist, gatelist, ids);
  // 2) LB loss
  moe_lb<<<1, 32, 0, stream>>>(gmean, loss);

  const size_t smem = (size_t)(TILE_M * D_ + TILE_M * H_ + 4 * 4096) * sizeof(__bf16);
  dim3 grid(T_ / TILE_M, E_);

  if (pre) {
    // 2b) one-shot fp32 -> bf16 transposed weight repack (LDS-tiled, coalesced)
    moe_repack<<<dim3(H_ / 32, D_ / 32, E_), 256, 0, stream>>>(w1, w1t, D_, H_);
    moe_repack<<<dim3(H_ / 32, D_ / 32, E_), 256, 0, stream>>>(w2, w2t, D_, H_);
    moe_repack<<<dim3(D_ / 32, H_ / 32, E_), 256, 0, stream>>>(wp, wpt, H_, D_);
    (void)hipFuncSetAttribute((const void*)moe_expert<true>,
                              hipFuncAttributeMaxDynamicSharedMemorySize, (int)smem);
    moe_expert<true><<<grid, 256, smem, stream>>>(x, xb, w1, w2, wp, w1t, w2t, wpt,
                                                  b1, b2, bp, counts, toklist,
                                                  gatelist, out);
  } else {
    (void)hipFuncSetAttribute((const void*)moe_expert<false>,
                              hipFuncAttributeMaxDynamicSharedMemorySize, (int)smem);
    moe_expert<false><<<grid, 256, smem, stream>>>(x, xb, w1, w2, wp, w1t, w2t, wpt,
                                                   b1, b2, bp, counts, toklist,
                                                   gatelist, out);
  }
}
